// BahdanauAttention_1039382085940
// MI455X (gfx1250) — matrix operand order repaired and checked by reading, not compiled, not run on hardware
//
#include <hip/hip_runtime.h>
#include <math.h>

#define B_ 64
#define S_ 4096
#define D_ 256
#define H_ 512
#define U_ 256
#define W1T_PITCH 264  // bf16 elems/row: 528B rows -> LDS bank stride 4 (conflict-free), 16B aligned

typedef __bf16 bf16;
typedef __attribute__((ext_vector_type(16))) bf16  v16bf;
typedef __attribute__((ext_vector_type(8)))  bf16  v8bf;
typedef __attribute__((ext_vector_type(8)))  float v8f;

// native fptrunc (RNE) -> v_cvt_pk_bf16_f32 on gfx1250 (confirmed in round-3 asm)
__device__ __forceinline__ bf16 f2bf(float f) { return (bf16)f; }

// Hardware v_tanh_f32 (confirmed in round-3 asm); branch-free fallback otherwise.
__device__ __forceinline__ float fast_tanh(float x) {
#if __has_builtin(__builtin_amdgcn_tanhf)
  return __builtin_amdgcn_tanhf(x);
#elif __has_builtin(__builtin_amdgcn_tanh_f32)
  return __builtin_amdgcn_tanh_f32(x);
#else
  float e = __builtin_amdgcn_exp2f(x * 2.8853900817779268f);  // e^(2x)
  return 1.f - 2.f * __builtin_amdgcn_rcpf(e + 1.f);
#endif
}

// exp for softmax; arguments are <= 0 after max subtraction, so v_exp_f32 is safe.
__device__ __forceinline__ float fast_exp(float x) {
  return __builtin_amdgcn_exp2f(x * 1.4426950408889634f);
}

// hb[b,u] = hidden[b,:] @ W2[:,u] + b2[u] + b1[u]   (both biases folded)
__global__ void __launch_bounds__(256) projh_kernel(
    const float* __restrict__ hidden, const float* __restrict__ W2,
    const float* __restrict__ b1, const float* __restrict__ b2,
    float* __restrict__ hb) {
  int t = blockIdx.x * 256 + threadIdx.x;  // 16384 = B*U
  int b = t >> 8, u = t & 255;
  const float* hrow = hidden + b * H_;
  float acc = b1[u] + b2[u];
  for (int h = 0; h < H_; ++h) acc = fmaf(hrow[h], W2[h * U_ + u], acc);
  hb[t] = acc;
}

// W1T[u][d] = bf16(W1[d][u]), padded pitch
__global__ void __launch_bounds__(256) w1t_kernel(const float* __restrict__ W1,
                                                  bf16* __restrict__ w1t) {
  int t = blockIdx.x * 256 + threadIdx.x;  // 65536 = D*U
  int u = t >> 8, d = t & 255;
  w1t[u * W1T_PITCH + d] = f2bf(W1[d * U_ + u]);
}

// B-fragment (32x16 bf16, ISA 7.12.2): lane n = lane&15, half g = lane>>4;
// element i holds K = (i/8)*16 + g*8 + (i%8) (+ 32*kb). Two 16B LDS loads, conflict-free via pitch.
__device__ __forceinline__ v16bf loadB(const bf16* __restrict__ sw, int u, int kb, int g) {
  const bf16* p = sw + u * W1T_PITCH + kb * 32 + g * 8;
  v8bf lo = *(const v8bf*)(p);
  v8bf hi = *(const v8bf*)(p + 16);
  v16bf bq;
#pragma unroll
  for (int j = 0; j < 8; ++j) { bq[j] = lo[j]; bq[j + 8] = hi[j]; }
  return bq;
}

// scores[b,s] = tanh(features[b,s,:]@W1 + hb[b,:]) @ V + bv
// grid (S/128, B), 256 threads = 8 waves, one 16-row tile per wave.
// (Round-2 proven shape: no spills; compiler interchanges loops and chains
//  128 WMMAs per tile against the LDS-resident W1T.)
__global__ void __launch_bounds__(256) score_kernel(
    const float* __restrict__ features, const bf16* __restrict__ w1t,
    const float* __restrict__ hb, const float* __restrict__ V,
    const float* __restrict__ bv, float* __restrict__ scores) {
  extern __shared__ char smemraw[];
  bf16* sw = (bf16*)smemraw;

  {  // cooperative fill of W1T (132 KB) into LDS
    const float4* src = (const float4*)w1t;
    float4* dst = (float4*)smemraw;
    const int n16 = (U_ * W1T_PITCH * 2) / 16;  // 8448
    for (int i = threadIdx.x; i < n16; i += 256) dst[i] = src[i];
  }
  __syncthreads();

  const int lane = threadIdx.x & 31;
  const int wave = threadIdx.x >> 5;
  const int nloc = lane & 15;  // N (and A-row m) within tile
  const int g    = lane >> 4;  // K-half select
  const int b    = blockIdx.y;
  const int s0   = blockIdx.x * 128 + wave * 16;

  // A fragments: 16-bit A 16x32 layout (ISA 7.12.2): lane m=nloc, half g;
  // element i holds K = (i/8)*16 + g*8 + (i%8)  (+ 32*kb)
  const float* arow = features + ((size_t)b * S_ + s0 + nloc) * D_;
  v16bf afrag[8];
#pragma unroll
  for (int kb = 0; kb < 8; ++kb) {
    const float* rp = arow + kb * 32 + g * 8;
    float4 x0 = *(const float4*)(rp);
    float4 x1 = *(const float4*)(rp + 4);
    float4 x2 = *(const float4*)(rp + 16);
    float4 x3 = *(const float4*)(rp + 20);
    v16bf a;
    a[0]  = f2bf(x0.x); a[1]  = f2bf(x0.y); a[2]  = f2bf(x0.z); a[3]  = f2bf(x0.w);
    a[4]  = f2bf(x1.x); a[5]  = f2bf(x1.y); a[6]  = f2bf(x1.z); a[7]  = f2bf(x1.w);
    a[8]  = f2bf(x2.x); a[9]  = f2bf(x2.y); a[10] = f2bf(x2.z); a[11] = f2bf(x2.w);
    a[12] = f2bf(x3.x); a[13] = f2bf(x3.y); a[14] = f2bf(x3.z); a[15] = f2bf(x3.w);
    afrag[kb] = a;
  }

  // hoist all per-column constants out of the hot loop
  const float* hbrow = hb + b * U_;
  float hbv[16], vnv[16];
#pragma unroll
  for (int nt = 0; nt < 16; ++nt) {
    hbv[nt] = hbrow[nt * 16 + nloc];
    vnv[nt] = V[nt * 16 + nloc];
  }

  float rowpart[8];
#pragma unroll
  for (int r = 0; r < 8; ++r) rowpart[r] = 0.f;

  for (int nt = 0; nt < 16; ++nt) {
    const int u = nt * 16 + nloc;
    v8f acc;  // C/D 16x16 f32: lane holds rows r+8*g, col nloc -> preload proj_h
#pragma unroll
    for (int r = 0; r < 8; ++r) acc[r] = hbv[nt];

    // software-pipelined B-frag loads: fetch kb+1 while WMMA kb executes
    v16bf bq = loadB(sw, u, 0, g);
#pragma unroll
    for (int kb = 0; kb < 8; ++kb) {
      v16bf bn = bq;
      if (kb < 7) bn = loadB(sw, u, kb + 1, g);
      acc = __builtin_amdgcn_wmma_f32_16x16x32_bf16(
          /*neg_a=*/false, afrag[kb], /*neg_b=*/false, bq,
          /*c_mod=*/(short)0, acc, /*reuse_a=*/false, /*reuse_b=*/false);
      bq = bn;
    }

    const float vn = vnv[nt];
#pragma unroll
    for (int r = 0; r < 8; ++r) rowpart[r] += fast_tanh(acc[r]) * vn;
  }

  // sum over N: butterfly across the 16 lanes of each half
#pragma unroll
  for (int r = 0; r < 8; ++r) {
    float v = rowpart[r];
    v += __shfl_xor(v, 1, 32);
    v += __shfl_xor(v, 2, 32);
    v += __shfl_xor(v, 4, 32);
    v += __shfl_xor(v, 8, 32);
    rowpart[r] = v;
  }
  if (nloc == 0) {
    const float bvv = bv[0];
#pragma unroll
    for (int r = 0; r < 8; ++r)
      scores[(size_t)b * S_ + s0 + g * 8 + r] = rowpart[r] + bvv;
  }
}

__global__ void __launch_bounds__(256) softmax_kernel(const float* __restrict__ scores,
                                                      float* __restrict__ wout) {
  __shared__ float red[256];
  const int b = blockIdx.x, t = threadIdx.x;
  const float* srow = scores + (size_t)b * S_;
  float m = -3.4e38f;
  for (int s = t; s < S_; s += 256) m = fmaxf(m, srow[s]);
  red[t] = m; __syncthreads();
  for (int o = 128; o > 0; o >>= 1) {
    if (t < o) red[t] = fmaxf(red[t], red[t + o]);
    __syncthreads();
  }
  const float mx = red[0];
  __syncthreads();
  float sum = 0.f;
  for (int s = t; s < S_; s += 256) sum += fast_exp(srow[s] - mx);
  red[t] = sum; __syncthreads();
  for (int o = 128; o > 0; o >>= 1) {
    if (t < o) red[t] += red[t + o];
    __syncthreads();
  }
  const float inv = 1.f / red[0];
  float* wrow = wout + (size_t)b * S_;
  for (int s = t; s < S_; s += 256) wrow[s] = fast_exp(srow[s] - mx) * inv;
}

// partial context sums (deterministic two-stage reduction, no atomics)
__global__ void __launch_bounds__(256) ctxpart_kernel(const float* __restrict__ features,
                                                      const float* __restrict__ w,
                                                      float* __restrict__ partials) {
  const int b = blockIdx.y, c = blockIdx.x, d = threadIdx.x;
  const int sbeg = c * (S_ / 8);
  const float* wrow = w + (size_t)b * S_ + sbeg;
  const float* f = features + ((size_t)b * S_ + sbeg) * D_ + d;
  float acc = 0.f;
  for (int s = 0; s < S_ / 8; ++s) {
    // speculative prefetch 24 rows (24 KB) ahead -> global_prefetch_b8
    __builtin_prefetch(f + (size_t)(s + 24) * D_, 0, 1);
    acc = fmaf(wrow[s], f[(size_t)s * D_], acc);
  }
  partials[((b * 8 + c) << 8) + d] = acc;
}

__global__ void __launch_bounds__(256) ctxreduce_kernel(const float* __restrict__ partials,
                                                        float* __restrict__ ctx) {
  int t = blockIdx.x * 256 + threadIdx.x;  // 16384 = B*D
  int b = t >> 8, d = t & 255;
  float acc = 0.f;
  for (int c = 0; c < 8; ++c) acc += partials[((b * 8 + c) << 8) + d];
  ctx[t] = acc;
}

extern "C" void kernel_launch(void* const* d_in, const int* in_sizes, int n_in,
                              void* d_out, int out_size, void* d_ws, size_t ws_size,
                              hipStream_t stream) {
  const float* features = (const float*)d_in[0];
  const float* hidden   = (const float*)d_in[1];
  const float* W1       = (const float*)d_in[2];
  const float* b1       = (const float*)d_in[3];
  const float* W2       = (const float*)d_in[4];
  const float* b2       = (const float*)d_in[5];
  const float* V        = (const float*)d_in[6];
  const float* bv       = (const float*)d_in[7];

  float* ctx_out  = (float*)d_out;             // [B, D]
  float* attn_out = (float*)d_out + B_ * D_;   // [B, S, 1]

  char* ws = (char*)d_ws;
  float* hb       = (float*)(ws);                               // 64 KB
  bf16*  w1t      = (bf16*)(ws + 65536);                        // 132 KB
  float* scores   = (float*)(ws + 65536 + 135168);              // 1 MB
  float* partials = (float*)(ws + 65536 + 135168 + 1048576);    // 512 KB

  projh_kernel<<<64, 256, 0, stream>>>(hidden, W2, b1, b2, hb);
  w1t_kernel<<<256, 256, 0, stream>>>(W1, w1t);

  const size_t lds = (size_t)U_ * W1T_PITCH * sizeof(bf16);  // 135168 B (<320 KB/WGP)
  hipFuncSetAttribute((const void*)score_kernel,
                      hipFuncAttributeMaxDynamicSharedMemorySize, (int)lds);
  score_kernel<<<dim3(S_ / 128, B_), 256, lds, stream>>>(features, w1t, hb, V, bv, scores);

  softmax_kernel<<<B_, 256, 0, stream>>>(scores, attn_out);
  ctxpart_kernel<<<dim3(8, B_), 256, 0, stream>>>(features, attn_out, partials);
  ctxreduce_kernel<<<64, 256, 0, stream>>>(partials, ctx_out);
}
